// NTXLoss_7370163880176
// MI455X (gfx1250) — matrix operand order repaired and checked by reading, not compiled, not run on hardware
//
#include <hip/hip_runtime.h>
#include <hip/hip_bf16.h>
#include <stdint.h>

// ---------- problem constants ----------
#define B_TOTAL 8192
#define NHALF   4096
#define DIM     2048
#define TEMP_INV 10.0f

// ---------- GEMM tiling ----------
#define BM 128
#define BN 128
#define KC 32
#define LDT 40                  // LDS row stride in halves (80B)
#define TILE_H (BM * LDT)       // halves per staged tile (5120)
#define TILE_B (TILE_H * 2)     // bytes per staged tile (10240)
#define BUF_H (4 * TILE_H)      // halves per k-chunk buffer (Ahi,Alo,Bhi,Blo)
#define SMEM_BYTES (2 * BUF_H * 2 + BM * 4)   // double buffer + rowsum

// ---------- WMMA / TDM types (gfx1250, wave32) ----------
typedef __attribute__((ext_vector_type(16))) __bf16          v16bf;
typedef __attribute__((ext_vector_type(16))) unsigned short  v16us;
typedef __attribute__((ext_vector_type(8)))  float           v8f;
typedef __attribute__((ext_vector_type(4)))  unsigned int    u32x4;
typedef __attribute__((ext_vector_type(8)))  int             i32x8;
typedef __attribute__((ext_vector_type(4)))  int             i32x4;

union FragU { uint4 q[2]; v16us u; };

#if __has_builtin(__builtin_amdgcn_tensor_load_to_lds) && \
    __has_builtin(__builtin_amdgcn_s_wait_tensorcnt)
#define USE_TDM 1
#else
#define USE_TDM 0
#endif

static __device__ __forceinline__ unsigned short f32_to_bf16_rn(float f) {
    uint32_t u = __float_as_uint(f);
    uint32_t r = u + 0x7FFFu + ((u >> 16) & 1u);
    return (unsigned short)(r >> 16);
}
static __device__ __forceinline__ float bf16_bits_to_f32(unsigned short h) {
    return __uint_as_float(((uint32_t)h) << 16);
}

// =====================================================================
// Kernel 1: row-normalize and split into bf16 hi/lo planes.
// =====================================================================
__global__ __launch_bounds__(256) void ntx_normalize_split(
        const float* __restrict__ x,
        unsigned short* __restrict__ xhi,
        unsigned short* __restrict__ xlo) {
    __shared__ float red[256];
    const int row = blockIdx.x;
    const int t   = threadIdx.x;
    const float4* xr = (const float4*)(x + (size_t)row * DIM);

    float4 a = xr[t];
    float4 b = xr[t + 256];
    float ss = a.x*a.x + a.y*a.y + a.z*a.z + a.w*a.w
             + b.x*b.x + b.y*b.y + b.z*b.z + b.w*b.w;
    red[t] = ss;
    __syncthreads();
    for (int s = 128; s > 0; s >>= 1) {
        if (t < s) red[t] += red[t + s];
        __syncthreads();
    }
    const float inv = 1.0f / fmaxf(sqrtf(red[0]), 1e-8f);

    float v[4];
    v[0]=a.x*inv; v[1]=a.y*inv; v[2]=a.z*inv; v[3]=a.w*inv;
    {
        unsigned short h0=f32_to_bf16_rn(v[0]), h1=f32_to_bf16_rn(v[1]),
                       h2=f32_to_bf16_rn(v[2]), h3=f32_to_bf16_rn(v[3]);
        ((ushort4*)(xhi + (size_t)row * DIM))[t] = make_ushort4(h0,h1,h2,h3);
        ((ushort4*)(xlo + (size_t)row * DIM))[t] =
            make_ushort4(f32_to_bf16_rn(v[0]-bf16_bits_to_f32(h0)),
                         f32_to_bf16_rn(v[1]-bf16_bits_to_f32(h1)),
                         f32_to_bf16_rn(v[2]-bf16_bits_to_f32(h2)),
                         f32_to_bf16_rn(v[3]-bf16_bits_to_f32(h3)));
    }
    v[0]=b.x*inv; v[1]=b.y*inv; v[2]=b.z*inv; v[3]=b.w*inv;
    {
        unsigned short h0=f32_to_bf16_rn(v[0]), h1=f32_to_bf16_rn(v[1]),
                       h2=f32_to_bf16_rn(v[2]), h3=f32_to_bf16_rn(v[3]);
        ((ushort4*)(xhi + (size_t)row * DIM))[t + 256] = make_ushort4(h0,h1,h2,h3);
        ((ushort4*)(xlo + (size_t)row * DIM))[t + 256] =
            make_ushort4(f32_to_bf16_rn(v[0]-bf16_bits_to_f32(h0)),
                         f32_to_bf16_rn(v[1]-bf16_bits_to_f32(h1)),
                         f32_to_bf16_rn(v[2]-bf16_bits_to_f32(h2)),
                         f32_to_bf16_rn(v[3]-bf16_bits_to_f32(h3)));
    }
}

// =====================================================================
// Kernel 2: zero the denominator accumulators.
// =====================================================================
__global__ void ntx_init(float* __restrict__ denom) {
    int i = blockIdx.x * blockDim.x + threadIdx.x;
    if (i < NHALF) denom[i] = 0.0f;
}

// =====================================================================
// TDM: load one 128x32 bf16 tile (row-major, tensor stride DIM) into LDS
// with 16B padding every 64B row -> 80B LDS row stride.
// D# layout per CDNA5 ISA. 6-arg builtin form
// (clang-23 / therock-10.0): (g0, g1, g2, g3, g4, cpol).
// =====================================================================
#if USE_TDM
static __device__ __forceinline__ void tdm_tile(const unsigned short* g,
                                                uint32_t ldsOff) {
    const uint64_t ga = (uint64_t)(uintptr_t)g;
    u32x4 g0;
    g0.x = 1u;                                          // count=1, no gather
    g0.y = ldsOff;                                      // lds_addr
    g0.z = (uint32_t)ga;                                // global_addr[31:0]
    g0.w = ((uint32_t)(ga >> 32) & 0x01FFFFFFu)         // global_addr[56:32]
         | (2u << 30);                                  // type=2 ("image")
    i32x8 g1;
    g1[0] = (1 << 16)        // data_size = 2 bytes
          | (1 << 20)        // pad_enable
          | (3 << 22)        // pad_interval: 16 DWORDs (one 64B tile row)
          | (3 << 25);       // pad_amount: 4 DWORDs (16B) -> 80B row stride
    g1[1] = (int)((unsigned)DIM << 16);      // tensor_dim0[15:0] @ bits 63:48
    g1[2] = (int)((unsigned)B_TOTAL << 16);  // tensor_dim1[15:0] @ bits 95:80
    g1[3] = (int)((unsigned)KC << 16);       // tile_dim0 @ bits 127:112
    g1[4] = BM;                              // tile_dim1 (tile_dim2 = 0)
    g1[5] = DIM;                             // tensor_dim0_stride low
    g1[6] = 0;
    g1[7] = 0;
    const i32x4 gz4 = {0, 0, 0, 0};
    const i32x8 gz8 = {0, 0, 0, 0, 0, 0, 0, 0};
    __builtin_amdgcn_tensor_load_to_lds(g0, g1, gz4, gz4, gz8, 0);
}

static __device__ __forceinline__ void stage_tdm(
        const unsigned short* xhi, const unsigned short* xlo,
        int rowBase, int colBase, int k, uint32_t ldsOff) {
    const size_t aOff = (size_t)rowBase * DIM + k;
    const size_t bOff = (size_t)colBase * DIM + k;
    tdm_tile(xhi + aOff, ldsOff + 0 * TILE_B);
    tdm_tile(xlo + aOff, ldsOff + 1 * TILE_B);
    tdm_tile(xhi + bOff, ldsOff + 2 * TILE_B);
    tdm_tile(xlo + bOff, ldsOff + 3 * TILE_B);
}
#else
static __device__ __forceinline__ void stage_coop(
        unsigned short* dst,                 // buffer base (4 tiles)
        const unsigned short* xhi, const unsigned short* xlo,
        int rowBase, int colBase, int k, int t) {
    #pragma unroll
    for (int c = 0; c < 2; ++c) {
        const int idx = t + c * 256;
        const int r   = idx >> 2;
        const int o   = (idx & 3) * 8;
        const size_t gA = (size_t)(rowBase + r) * DIM + k + o;
        const size_t gB = (size_t)(colBase + r) * DIM + k + o;
        *(uint4*)&dst[0 * TILE_H + r * LDT + o] = *(const uint4*)&xhi[gA];
        *(uint4*)&dst[1 * TILE_H + r * LDT + o] = *(const uint4*)&xlo[gA];
        *(uint4*)&dst[2 * TILE_H + r * LDT + o] = *(const uint4*)&xhi[gB];
        *(uint4*)&dst[3 * TILE_H + r * LDT + o] = *(const uint4*)&xlo[gB];
    }
}
#endif

// =====================================================================
// Kernel 3: fused GEMM + exp row-sum epilogue, TDM double-buffered.
// sim = hi*hi + hi*lo + lo*hi  (3x bf16 split, ~f32 accuracy)
// =====================================================================
__global__ __launch_bounds__(256) void ntx_gemm_fused(
        const unsigned short* __restrict__ xhi,
        const unsigned short* __restrict__ xlo,
        float* __restrict__ denom,
        float* __restrict__ posSim) {
    extern __shared__ unsigned short smem[];             // 2 buffers of 4 tiles
    float* rowsum = (float*)(smem + 2 * BUF_H);

    const int bc = blockIdx.x;            // 0..63  -> cols 0..8191
    const int br = blockIdx.y;            // 0..31  -> rows 0..4095
    const int rowBase = br * BM;
    const int colBase = bc * BN;

    const int t    = threadIdx.x;
    const int lane = t & 31;
    const int wave = t >> 5;
    const int waveRow = wave & 3;         // 32-row slab
    const int waveCol = wave >> 2;        // 64-col slab
    const int m16  = lane & 15;
    const int hsel = lane >> 4;

    if (t < BM) rowsum[t] = 0.0f;

    v8f acc[2][4];
    #pragma unroll
    for (int mt = 0; mt < 2; ++mt)
        #pragma unroll
        for (int nt = 0; nt < 4; ++nt)
            acc[mt][nt] = (v8f){0,0,0,0,0,0,0,0};

#if USE_TDM
    const uint32_t ldsBase = (uint32_t)(uintptr_t)(void*)smem;
    if (wave == 0) stage_tdm(xhi, xlo, rowBase, colBase, 0, ldsBase);
#else
    stage_coop(smem, xhi, xlo, rowBase, colBase, 0, t);
#endif

    const int NK = DIM / KC;
    for (int kt = 0; kt < NK; ++kt) {
        const int buf = kt & 1;
#if USE_TDM
        if (wave == 0) __builtin_amdgcn_s_wait_tensorcnt(0);
#endif
        __syncthreads();   // tile kt resident; everyone done with buf^1's data
        if (kt + 1 < NK) {
#if USE_TDM
            if (wave == 0)
                stage_tdm(xhi, xlo, rowBase, colBase, (kt + 1) * KC,
                          ldsBase + (uint32_t)(buf ^ 1) * (BUF_H * 2));
#else
            stage_coop(smem + (buf ^ 1) * BUF_H, xhi, xlo,
                       rowBase, colBase, (kt + 1) * KC, t);
#endif
        }

        const unsigned short* sAhi = smem + buf * BUF_H;
        const unsigned short* sAlo = sAhi + TILE_H;
        const unsigned short* sBhi = sAlo + TILE_H;
        const unsigned short* sBlo = sBhi + TILE_H;

        // A fragments: lane m=lane%16, halves [8h,8h+8) U [16+8h,16+8h+8)
        FragU aHi[2], aLo[2];
        #pragma unroll
        for (int mt = 0; mt < 2; ++mt) {
            const int base = (waveRow * 32 + mt * 16 + m16) * LDT;
            aHi[mt].q[0] = *(const uint4*)&sAhi[base + 8 * hsel];
            aHi[mt].q[1] = *(const uint4*)&sAhi[base + 16 + 8 * hsel];
            aLo[mt].q[0] = *(const uint4*)&sAlo[base + 8 * hsel];
            aLo[mt].q[1] = *(const uint4*)&sAlo[base + 16 + 8 * hsel];
        }
        // B fragments: lane n=lane%16, halves [8h,8h+8) U [16+8h,16+8h+8)
        FragU bHi[4], bLo[4];
        #pragma unroll
        for (int nt = 0; nt < 4; ++nt) {
            const int base = (waveCol * 64 + nt * 16 + m16) * LDT;
            bHi[nt].q[0] = *(const uint4*)&sBhi[base + 8 * hsel];
            bHi[nt].q[1] = *(const uint4*)&sBhi[base + 16 + 8 * hsel];
            bLo[nt].q[0] = *(const uint4*)&sBlo[base + 8 * hsel];
            bLo[nt].q[1] = *(const uint4*)&sBlo[base + 16 + 8 * hsel];
        }

        #pragma unroll
        for (int mt = 0; mt < 2; ++mt) {
            const v16bf ah = __builtin_bit_cast(v16bf, aHi[mt].u);
            const v16bf al = __builtin_bit_cast(v16bf, aLo[mt].u);
            #pragma unroll
            for (int nt = 0; nt < 4; ++nt) {
                const v16bf bh = __builtin_bit_cast(v16bf, bHi[nt].u);
                const v16bf bl = __builtin_bit_cast(v16bf, bLo[nt].u);
                acc[mt][nt] = __builtin_amdgcn_wmma_f32_16x16x32_bf16(
                    false, ah, false, bh, (short)0, acc[mt][nt], false, false);
                acc[mt][nt] = __builtin_amdgcn_wmma_f32_16x16x32_bf16(
                    false, ah, false, bl, (short)0, acc[mt][nt], false, false);
                acc[mt][nt] = __builtin_amdgcn_wmma_f32_16x16x32_bf16(
                    false, al, false, bh, (short)0, acc[mt][nt], false, false);
            }
        }
    }

    // ---- epilogue: e = exp(10*sim); drop diagonal; capture positive pair ----
    // C/D layout: lane l, VGPR v -> M = v + 8*(l>>4), N = l%16
    const int mrow = hsel * 8;
    const int ncol = m16;
    #pragma unroll
    for (int mt = 0; mt < 2; ++mt) {
        const int lRowBase = waveRow * 32 + mt * 16;
        #pragma unroll
        for (int nt = 0; nt < 4; ++nt) {
            const int gCol = colBase + waveCol * 64 + nt * 16 + ncol;
            #pragma unroll
            for (int v = 0; v < 8; ++v) {
                const int lRow = lRowBase + mrow + v;
                const int gRow = rowBase + lRow;
                const float s  = acc[mt][nt][v];
                float e = __expf(s * TEMP_INV);
                if (gCol == gRow) e = 0.0f;                 // self term removed
                if (gCol == gRow + NHALF) posSim[gRow] = s; // positive pair
                e += __shfl_xor(e, 1, 32);
                e += __shfl_xor(e, 2, 32);
                e += __shfl_xor(e, 4, 32);
                e += __shfl_xor(e, 8, 32);
                if (ncol == 0) atomicAdd(&rowsum[lRow], e);
            }
        }
    }
    __syncthreads();
    if (t < BM) atomicAdd(&denom[rowBase + t], rowsum[t]);
}

// =====================================================================
// Kernel 4: loss = sum_i<4096 (log(denom[i]) - 10*posSim[i]) / 8192
// =====================================================================
__global__ __launch_bounds__(256) void ntx_finalize(
        const float* __restrict__ denom,
        const float* __restrict__ posSim,
        float* __restrict__ out) {
    __shared__ float red[256];
    const int t = threadIdx.x;
    float p = 0.0f;
    for (int i = t; i < NHALF; i += 256)
        p += logf(denom[i]) - posSim[i] * TEMP_INV;
    red[t] = p;
    __syncthreads();
    for (int s = 128; s > 0; s >>= 1) {
        if (t < s) red[t] += red[t + s];
        __syncthreads();
    }
    if (t == 0) out[0] = red[0] / (float)B_TOTAL;
}

// =====================================================================
extern "C" void kernel_launch(void* const* d_in, const int* in_sizes, int n_in,
                              void* d_out, int out_size, void* d_ws, size_t ws_size,
                              hipStream_t stream) {
    (void)in_sizes; (void)n_in; (void)out_size; (void)ws_size;
    const float* x = (const float*)d_in[0];
    float* out = (float*)d_out;

    unsigned short* xhi = (unsigned short*)d_ws;                 // 8192*2048 u16
    unsigned short* xlo = xhi + (size_t)B_TOTAL * DIM;           // 8192*2048 u16
    float* denom  = (float*)(xlo + (size_t)B_TOTAL * DIM);       // 4096 f32
    float* posSim = denom + NHALF;                               // 4096 f32

    ntx_normalize_split<<<B_TOTAL, 256, 0, stream>>>(x, xhi, xlo);
    ntx_init<<<(NHALF + 255) / 256, 256, 0, stream>>>(denom);
    ntx_gemm_fused<<<dim3(B_TOTAL / BN, NHALF / BM), 256, SMEM_BYTES, stream>>>(
        xhi, xlo, denom, posSim);
    ntx_finalize<<<1, 256, 0, stream>>>(denom, posSim, out);
}
